// SingleHeadAttention_31112743092586
// MI455X (gfx1250) — matrix operand order repaired and checked by reading, not compiled, not run on hardware
//
#include <hip/hip_runtime.h>
#include <hip/hip_bf16.h>

// ---------------------------------------------------------------------------
// SingleHeadAttention over 4 tokens, d=1024, batch=16384 — MI455X (gfx1250).
//
// Algebraic refactor (exact):
//   scores = (xWq^T+bq)(xWk^T+bk)^T / 32
//          = [ x (Wk^T Wq)^T x^T + x·a + b·x + s0 ] / 32,
//              P = Wk^T Wq (so T = X P^T is a plain Linear),
//              a = Wq^T bk, b = Wk^T bq, s0 = bq·bk
//   fused  = (c^T x) Wv^T + bv,  c_m = sum_n pool_n attn[n,m]
// => two WMMA GEMM passes (X P^T, Y Wv^T); Q/K/V never materialized in HBM.
// HBM traffic = x once (256 MB) + out (64 MB) ~= 14 us floor @ 23.3 TB/s.
// ---------------------------------------------------------------------------

#define D      1024
#define NIMG   4
#define NBATCH 16384
#define ROWS   (NBATCH * NIMG)   // 65536 flattened (batch,image) rows

#define PH 1032   // LDS tile pitch, 16-bit units (1024 + 8 pad; 516 dwords == 4 mod 64 banks)
#define PF 516    // same byte pitch in 32-bit units

typedef __attribute__((ext_vector_type(16))) _Float16 v16h;
typedef __attribute__((ext_vector_type(8)))  _Float16 v8h;
typedef __attribute__((ext_vector_type(4)))  _Float16 v4h;
typedef __attribute__((ext_vector_type(8)))  float    v8f;

static __device__ __forceinline__ v16h combine16(v8h lo, v8h hi) {
  v16h r;
#pragma unroll
  for (int i = 0; i < 8; ++i) { r[i] = lo[i]; r[i + 8] = hi[i]; }
  return r;
}

static __device__ __forceinline__ v8f wmma_f32_f16(v16h a, v16h b, v8f c) {
  // D(16x16 f32) = A(16x32 f16) x B(32x16 f16) + C
  return __builtin_amdgcn_wmma_f32_16x16x32_f16(false, a, false, b, (short)0, c,
                                                false, false);
}

// ---------------------------------------------------------------------------
// Kernel 0a: bias cross-terms  a = Wq^T bk, b = Wk^T bq, s0 = bq.bk
// (bq=bk=0 in the harness, but kept for generality; negligible cost)
// ---------------------------------------------------------------------------
__global__ __launch_bounds__(256) void k_prep(const float* __restrict__ Wq,
                                              const float* __restrict__ Wk,
                                              const float* __restrict__ bq,
                                              const float* __restrict__ bk,
                                              float* __restrict__ a_out,
                                              float* __restrict__ b_out,
                                              float* __restrict__ s0_out) {
  const int d = blockIdx.x * 256 + threadIdx.x;   // grid = D/256, coalesced columns
  float sa = 0.f, sb = 0.f;
  for (int e = 0; e < D; ++e) {
    sa += Wq[(size_t)e * D + d] * bk[e];
    sb += Wk[(size_t)e * D + d] * bq[e];
  }
  a_out[d] = sa;
  b_out[d] = sb;
  if (d == 0) {
    float s = 0.f;
    for (int e = 0; e < D; ++e) s += bq[e] * bk[e];
    s0_out[0] = s;
  }
}

// ---------------------------------------------------------------------------
// Kernel 0b: P[r][c] = sum_e Wk[e][r] * Wq[e][c]   (1024x1024, f16 out to ws)
// WG tile 64(r) x 128(c), 8 waves; A/B staged through LDS in f16.
// ---------------------------------------------------------------------------
__global__ __launch_bounds__(256) void k_gemmP(const float* __restrict__ Wk,
                                               const float* __restrict__ Wq,
                                               _Float16* __restrict__ P) {
  __shared__ _Float16 Ah[64 * 40];    // [r][k], pitch 40 halves (bank-safe)
  __shared__ _Float16 Bh[128 * 40];   // [c][k]

  const int tid = threadIdx.x;
  const int wid = tid >> 5, lane = tid & 31;
  const int nn = lane & 15, kh = lane >> 4;
  const int r0 = (blockIdx.x >> 3) * 64;
  const int c0 = (blockIdx.x & 7) * 128;

  v8f acc[4] = {};
  for (int k0 = 0; k0 < D; k0 += 32) {
    // stage A slice: Wk[k0..k0+31][r0..r0+63] -> Ah[r][k] (coalesced global reads)
    for (int i = tid; i < 64 * 32; i += 256) {
      int k = i >> 6, r = i & 63;
      Ah[r * 40 + k] = (_Float16)Wk[(size_t)(k0 + k) * D + r0 + r];
    }
    // stage B slice: Wq[k0..k0+31][c0..c0+127] -> Bh[c][k]
    for (int i = tid; i < 128 * 32; i += 256) {
      int k = i >> 7, c = i & 127;
      Bh[c * 40 + k] = (_Float16)Wq[(size_t)(k0 + k) * D + c0 + c];
    }
    __syncthreads();

    // B fragment: lane half kh holds 16 contiguous K at column (wid*16+nn)
    const _Float16* bp = &Bh[(wid * 16 + nn) * 40 + kh * 16];
    v16h bf = combine16(*(const v8h*)bp, *(const v8h*)(bp + 8));
#pragma unroll
    for (int mt = 0; mt < 4; ++mt) {
      const _Float16* ap = &Ah[(mt * 16 + nn) * 40];
      v16h af = combine16(*(const v8h*)(ap + kh * 16), *(const v8h*)(ap + kh * 16 + 8));
      acc[mt] = wmma_f32_f16(af, bf, acc[mt]);
    }
    __syncthreads();
  }
#pragma unroll
  for (int mt = 0; mt < 4; ++mt)
#pragma unroll
    for (int v = 0; v < 8; ++v) {
      int row = r0 + mt * 16 + v + 8 * kh;      // C/D layout: M = v + 8*(lane>=16)
      int col = c0 + wid * 16 + nn;             // N = lane%16
      P[(size_t)row * D + col] = (_Float16)acc[mt][v];
    }
}

// ---------------------------------------------------------------------------
// Kernel 1: fully fused attention. One WG = 64 rows (16 batches), 256 thr.
// LDS phases: Xh(f16 64xD) | Th(f16 64xD) | Yh(f16 16xD) | scratch
//   Th is re-aliased as res(f32 16xD) after scores; Xh as pre(f32 16xD) after Y.
// 298 KB LDS of the 320 KB/WGP -> single resident WG, 8 wave32s.
// ---------------------------------------------------------------------------
__global__ __launch_bounds__(256, 1) void k_main(
    const float* __restrict__ x, const _Float16* __restrict__ P,
    const float* __restrict__ a_g, const float* __restrict__ b_g,
    const float* __restrict__ s0_g, const float* __restrict__ Wv,
    const float* __restrict__ bv, const float* __restrict__ gamma,
    const float* __restrict__ beta, const float* __restrict__ pool,
    float* __restrict__ out) {
  extern __shared__ char smem[];
  _Float16* Xh = (_Float16*)smem;      // 64 x PH halves
  _Float16* Th = Xh + 64 * PH;         // 64 x PH halves
  _Float16* Yh = Th + 64 * PH;         // 16 x PH halves
  float* scr = (float*)(Yh + 16 * PH); // u[64] | w[64] | cc[64]
  float* u  = scr;
  float* w  = scr + 64;
  float* cc = scr + 128;
  float* res = (float*)Th;             // alias (valid after scores barrier)
  float* pre = (float*)Xh;             // alias (valid after phase-Y barrier)

  const int tid  = threadIdx.x;
  const int wid  = tid >> 5, lane = tid & 31;
  const int nn   = lane & 15, kh = lane >> 4;
  const int r0g  = blockIdx.x * 64;    // flattened (batch,image) row base

  if (tid < 192) scr[tid] = 0.f;

  // ---- load X tile: 64 rows x 1024 f32 -> f16 LDS (float4 in, v4h out) ----
  for (int i = tid; i < 64 * 256; i += 256) {
    int row = i >> 8, c4 = (i & 255) << 2;
    const float4 v = ((const float4*)(x + (size_t)(r0g + row) * D))[i & 255];
    v4h hv;
    hv[0] = (_Float16)v.x; hv[1] = (_Float16)v.y;
    hv[2] = (_Float16)v.z; hv[3] = (_Float16)v.w;
    *(v4h*)(Xh + row * PH + c4) = hv;            // single ds_store_b64
  }
  __syncthreads();

  // ---- bias cross-terms per row: u[r]=x_r·a, w[r]=b·x_r (4 partials/row) ----
  {
    int row = tid & 63, q = tid >> 6;
    float su = 0.f, sw = 0.f;
    for (int d = q * 256; d < q * 256 + 256; d += 4) {
      v4h    xv = *(const v4h*)&Xh[row * PH + d];
      float4 av = *(const float4*)&a_g[d];
      float4 bw = *(const float4*)&b_g[d];
      su += (float)xv[0] * av.x + (float)xv[1] * av.y +
            (float)xv[2] * av.z + (float)xv[3] * av.w;
      sw += (float)xv[0] * bw.x + (float)xv[1] * bw.y +
            (float)xv[2] * bw.z + (float)xv[3] * bw.w;
    }
    atomicAdd(&u[row], su);   // ds_add_f32
    atomicAdd(&w[row], sw);
  }

  // ---- phase T: T = X @ P^T (64x1024, K=1024) via v_wmma_f32_16x16x32_f16 ----
  for (int nt = 0; nt < 8; ++nt) {
    const int col0 = wid * 128 + nt * 16;
    v8f acc[4] = {};
    const _Float16* pb_row = P + (size_t)(col0 + nn) * D;  // B[k][n] = P[col0+n][k]
    for (int k0 = 0; k0 < D; k0 += 32) {
      const _Float16* pb = pb_row + k0 + kh * 16;
      __builtin_prefetch(pb + 32, 0, 1);                   // global_prefetch_b8
      v16h bf = combine16(*(const v8h*)pb, *(const v8h*)(pb + 8));
#pragma unroll
      for (int mt = 0; mt < 4; ++mt) {
        const _Float16* ap = Xh + (mt * 16 + nn) * PH + k0;
        v16h af = combine16(*(const v8h*)(ap + kh * 16), *(const v8h*)(ap + kh * 16 + 8));
        acc[mt] = wmma_f32_f16(af, bf, acc[mt]);
      }
    }
#pragma unroll
    for (int mt = 0; mt < 4; ++mt)
#pragma unroll
      for (int v = 0; v < 8; ++v)
        Th[(mt * 16 + v + 8 * kh) * PH + col0 + nn] = (_Float16)acc[mt][v];
  }
  __syncthreads();

  // ---- scores -> softmax -> pooled coefficients c (2 batches per wave) ----
  {
    const int grp = lane >> 4;         // which batch of this wave's pair
    const int idx = lane & 15;
    const int sn = idx >> 2, sm = idx & 3;
    const int bb = wid * 2 + grp;      // local batch 0..15
    const int rn = bb * 4 + sn, rm = bb * 4 + sm;
    float dot = 0.f;
    for (int d0 = 0; d0 < D; d0 += 8) {          // v8h: ds_load_b128 pairs
      v8h tv = *(const v8h*)&Th[rn * PH + d0];
      v8h xv = *(const v8h*)&Xh[rm * PH + d0];
#pragma unroll
      for (int j = 0; j < 8; ++j) dot += (float)tv[j] * (float)xv[j];
    }
    float s = 0.03125f * (dot + u[rn] + w[rm] + s0_g[0]);  // 1/sqrt(1024)
    // softmax over m within each quartet of lanes
    float mx = fmaxf(s, __shfl_xor(s, 1));
    mx = fmaxf(mx, __shfl_xor(mx, 2));
    float e = __expf(s - mx);
    float den = e + __shfl_xor(e, 1);
    den += __shfl_xor(den, 2);
    float attn = e / den;
    // pool softmax (4 elements, per lane)
    float p0 = pool[0], p1 = pool[1], p2 = pool[2], p3 = pool[3];
    float pm = fmaxf(fmaxf(p0, p1), fmaxf(p2, p3));
    float e0 = __expf(p0 - pm), e1 = __expf(p1 - pm);
    float e2 = __expf(p2 - pm), e3 = __expf(p3 - pm);
    float psum = e0 + e1 + e2 + e3;
    float pwn = (sn == 0 ? e0 : sn == 1 ? e1 : sn == 2 ? e2 : e3) / psum;
    // c_m = sum_n pool_n * attn[n][m]  (reduce over n-lanes)
    float val = pwn * attn;
    val += __shfl_xor(val, 4);
    val += __shfl_xor(val, 8);
    if (idx < 4) cc[bb * 4 + sm] = val;   // idx==m when n==0
  }
  __syncthreads();

  // ---- phase Y: y_b = sum_m c_m x_m ; residual = mean_m x_m (res over Th) ----
  for (int i = tid; i < 16 * 256; i += 256) {
    int bb = i >> 8, d4 = (i & 255) << 2;
    float c0 = cc[bb * 4 + 0], c1 = cc[bb * 4 + 1];
    float c2 = cc[bb * 4 + 2], c3 = cc[bb * 4 + 3];
    v4h x0 = *(const v4h*)&Xh[(bb * 4 + 0) * PH + d4];
    v4h x1 = *(const v4h*)&Xh[(bb * 4 + 1) * PH + d4];
    v4h x2 = *(const v4h*)&Xh[(bb * 4 + 2) * PH + d4];
    v4h x3 = *(const v4h*)&Xh[(bb * 4 + 3) * PH + d4];
    v4h yv;
    float rr[4];
#pragma unroll
    for (int j = 0; j < 4; ++j) {
      float a0 = (float)x0[j], a1 = (float)x1[j];
      float a2 = (float)x2[j], a3 = (float)x3[j];
      yv[j] = (_Float16)(c0 * a0 + c1 * a1 + c2 * a2 + c3 * a3);
      rr[j] = 0.25f * (a0 + a1 + a2 + a3);
    }
    *(v4h*)&Yh[bb * PH + d4] = yv;
    *(float4*)&res[bb * PF + d4] = *(float4*)rr;
  }
  __syncthreads();

  // ---- phase B: pre = Y @ Wv^T + bv + res   (16x1024, K=1024, WMMA) ----
  for (int nt = 0; nt < 8; ++nt) {
    const int col0 = wid * 128 + nt * 16;
    v8f acc = {};
    const float* wv_row = Wv + (size_t)(col0 + nn) * D;    // B[k][n] = Wv[col0+n][k]
    for (int k0 = 0; k0 < D; k0 += 32) {
      const float* bp = wv_row + k0 + kh * 16;
      __builtin_prefetch(bp + 32, 0, 1);
      v16h bf;
#pragma unroll
      for (int j = 0; j < 16; ++j) bf[j] = (_Float16)bp[j];
      const _Float16* ap = Yh + nn * PH + k0;
      v16h af = combine16(*(const v8h*)(ap + kh * 16), *(const v8h*)(ap + kh * 16 + 8));
      acc = wmma_f32_f16(af, bf, acc);
    }
#pragma unroll
    for (int v = 0; v < 8; ++v) {
      int row = v + 8 * kh;            // batch 0..15
      int col = col0 + nn;
      pre[row * PF + col] = acc[v] + bv[col] + res[row * PF + col];
    }
  }
  __syncthreads();

  // ---- LayerNorm + store (2 batches per wave, lane-parallel over d) ----
  for (int bi = 0; bi < 2; ++bi) {
    const int row = wid * 2 + bi;
    const size_t gb = (size_t)blockIdx.x * 16 + row;
    float sum = 0.f, ss = 0.f;
    for (int d = lane; d < D; d += 32) {
      float v = pre[row * PF + d];
      sum += v; ss += v * v;
    }
#pragma unroll
    for (int off = 16; off > 0; off >>= 1) {
      sum += __shfl_xor(sum, off);
      ss  += __shfl_xor(ss, off);
    }
    float mu = sum * (1.f / D);
    float var = ss * (1.f / D) - mu * mu;
    float rs = rsqrtf(var + 1e-5f);
    for (int d = lane; d < D; d += 32)
      out[gb * D + d] = (pre[row * PF + d] - mu) * rs * gamma[d] + beta[d];
  }
}

// ---------------------------------------------------------------------------
extern "C" void kernel_launch(void* const* d_in, const int* in_sizes, int n_in,
                              void* d_out, int out_size, void* d_ws, size_t ws_size,
                              hipStream_t stream) {
  const float* x     = (const float*)d_in[0];
  const float* Wq    = (const float*)d_in[1];
  const float* bq    = (const float*)d_in[2];
  const float* Wk    = (const float*)d_in[3];
  const float* bk    = (const float*)d_in[4];
  const float* Wv    = (const float*)d_in[5];
  const float* bv    = (const float*)d_in[6];
  const float* gamma = (const float*)d_in[7];
  const float* beta  = (const float*)d_in[8];
  const float* pool  = (const float*)d_in[9];
  float* out = (float*)d_out;

  // workspace layout: P (f16, 2 MB) | a (4 KB) | b (4 KB) | s0 (4 B)
  _Float16* P = (_Float16*)d_ws;
  float* a_g  = (float*)((char*)d_ws + (size_t)D * D * sizeof(_Float16));
  float* b_g  = a_g + D;
  float* s0_g = b_g + D;

  constexpr size_t SMEM_BYTES = (size_t)(64 + 64 + 16) * PH * sizeof(_Float16) + 1024;
  hipFuncSetAttribute((const void*)k_main,
                      hipFuncAttributeMaxDynamicSharedMemorySize, (int)SMEM_BYTES);

  k_prep <<<D / 256, 256, 0, stream>>>(Wq, Wk, bq, bk, a_g, b_g, s0_g);
  k_gemmP<<<(D / 64) * (D / 128), 256, 0, stream>>>(Wk, Wq, P);
  k_main <<<ROWS / 64, 256, SMEM_BYTES, stream>>>(x, P, a_g, b_g, s0_g, Wv, bv,
                                                  gamma, beta, pool, out);
}
